// CausalAttention_27788438405181
// MI455X (gfx1250) — compile-verified, matrix operand order checked by reading, not run on hardware
//
#include <hip/hip_runtime.h>

#define Bd 4
#define Sd 2048
#define Dd 1024
#define Hd 16
#define DKd 64
#define Md (Bd * Sd)  // 8192 rows

typedef __attribute__((ext_vector_type(16))) __bf16 v16bf;
typedef __attribute__((ext_vector_type(8)))  float  v8f;
typedef __attribute__((ext_vector_type(4)))  int    v4i;

#if __has_builtin(__builtin_amdgcn_global_load_async_to_lds_b128) && \
    __has_builtin(__builtin_amdgcn_s_wait_asynccnt)
#define ASYNC_LDS 1
#endif

__device__ __forceinline__ __bf16 f2bf(float f) { return (__bf16)f; }

// Copy 16 bytes global -> LDS. Async (DMA, ASYNCcnt-tracked) when available,
// else a synchronous vector copy through VGPRs.
__device__ __forceinline__ void cp16_g2l(__bf16* lds, const __bf16* glob) {
#if defined(ASYNC_LDS)
  typedef __attribute__((address_space(1))) v4i* gptr_t;  // global v4i*
  typedef __attribute__((address_space(3))) v4i* lptr_t;  // LDS v4i*
  __builtin_amdgcn_global_load_async_to_lds_b128((gptr_t)glob, (lptr_t)lds, 0, 0);
#else
  *(v4i*)lds = *(const v4i*)glob;
#endif
}

__device__ __forceinline__ void wait_async_copies() {
#if defined(ASYNC_LDS)
  __builtin_amdgcn_s_wait_asynccnt(0);
#endif
}

// Load a 16x32 bf16 WMMA fragment (A-layout; B 32x16 uses the identical per-lane
// pattern with "row" = output column, since both sources are K-contiguous).
__device__ __forceinline__ v16bf load_frag16(const __bf16* __restrict__ base,
                                             int row, int ld, int half) {
  const __bf16* p = base + (size_t)row * ld;
  v16bf a;
#pragma unroll
  for (int v = 0; v < 8; ++v) {
    int k = ((v & 4) ? 16 : 0) + half * 8 + (v & 3) * 2;
    a[2 * v]     = p[k];
    a[2 * v + 1] = p[k + 1];
  }
  return a;
}

// ---------------- f32 -> bf16 conversion ----------------
__global__ __launch_bounds__(256) void cvt_f32_bf16_kernel(
    const float* __restrict__ in, __bf16* __restrict__ out, int n) {
  int i = blockIdx.x * blockDim.x + threadIdx.x;
  int stride = gridDim.x * blockDim.x;
  for (; i < n; i += stride) out[i] = f2bf(in[i]);
}

// ---------------- QKV projection: dst = X @ W^T, head-split store ----------------
// All 8 waves of a block share one 16-col B tile; its 16x32 k-slices are staged
// in LDS (double-buffered async) and reused 8x.
__global__ __launch_bounds__(256) void qkv_gemm_kernel(
    const __bf16* __restrict__ X, const __bf16* __restrict__ W,
    __bf16* __restrict__ dst, int mode) {
  __shared__ __bf16 bsh[2][16][32];  // double-buffered B k-slice (1 KiB each)
  int tid  = threadIdx.x;
  int lane = tid & 31;
  int wid  = tid >> 5;
  int half = lane >> 4;
  int lq   = lane & 15;
  int mt = blockIdx.y * 8 + wid;  // 512 row tiles
  int nt = blockIdx.x;            // 64 col tiles

  const __bf16* arow = X + (size_t)(mt * 16) * Dd;
  const __bf16* brow = W + (size_t)(nt * 16) * Dd;

  // prologue: stage k-slice 0 (64 lanes x 16B = 1 KiB)
  if (tid < 64)
    cp16_g2l(&bsh[0][tid >> 2][(tid & 3) * 8], brow + (size_t)(tid >> 2) * Dd + (tid & 3) * 8);

  v8f acc = {};
  for (int kk = 0; kk < Dd; kk += 32) {
    int cur = (kk >> 5) & 1;
    wait_async_copies();
    __syncthreads();
    if (kk + 32 < Dd && tid < 64)
      cp16_g2l(&bsh[cur ^ 1][tid >> 2][(tid & 3) * 8],
               brow + kk + 32 + (size_t)(tid >> 2) * Dd + (tid & 3) * 8);
    __builtin_prefetch(arow + kk + 128, 0, 3);
    v16bf a = load_frag16(arow + kk, lq, Dd, half);
    v16bf b = load_frag16(&bsh[cur][0][0], lq, 32, half);
    acc = __builtin_amdgcn_wmma_f32_16x16x32_bf16(false, a, false, b,
                                                  (short)0, acc, false, false);
    __syncthreads();
  }
#pragma unroll
  for (int r = 0; r < 8; ++r) {
    int m = mt * 16 + r + half * 8;
    int n = nt * 16 + lq;
    int bb = m / Sd, s = m % Sd;
    int h = n / DKd, dk = n % DKd;
    __bf16 val = f2bf(acc[r]);
    if (mode == 2) {
      dst[((size_t)(bb * Hd + h) * DKd + dk) * Sd + s] = val;  // V^T
    } else {
      dst[((size_t)(bb * Hd + h) * Sd + s) * DKd + dk] = val;  // Q, K
    }
  }
}

// ---------------- Flash attention ----------------
// One wave per 16-query tile; the 8 waves of a block share (b,h) and walk the
// same key blocks, so K (32x64) and V^T (64x32) tiles are staged in LDS
// (double-buffered async) and reused by all 8 waves. Causal divergence is
// handled by looping to the block-max key block with uniform barriers.
__global__ __launch_bounds__(256) void flash_attn_kernel(
    const __bf16* __restrict__ Q,   // [B,H,S,DK]
    const __bf16* __restrict__ Km,  // [B,H,S,DK]
    const __bf16* __restrict__ Vt,  // [B,H,DK,S]
    __bf16* __restrict__ O) {       // [B,S,D] bf16
  __shared__ __bf16 ksh[2][32][64];   // keys x dk      (4 KiB each)
  __shared__ __bf16 vsh[2][64][32];   // dk   x keys    (4 KiB each)
  __shared__ __bf16 pbuf[8][16][32];  // per-wave P scratch (C-layout -> A-layout)

  int tid  = threadIdx.x;
  int lane = tid & 31;
  int wid  = tid >> 5;
  int half = lane >> 4;
  int lq   = lane & 15;
  int bh    = blockIdx.x >> 4;          // 16 blocks per (b,h)
  int qbase = (blockIdx.x & 15) * 8;    // first q-tile of this block
  int qt    = qbase + wid;

  const __bf16* Qb = Q  + (size_t)bh * Sd * DKd;
  const __bf16* Kb = Km + (size_t)bh * Sd * DKd;
  const __bf16* Vb = Vt + (size_t)bh * DKd * Sd;

  v16bf qa0 = load_frag16(Qb + qt * 16 * DKd + 0,  lq, DKd, half);
  v16bf qa1 = load_frag16(Qb + qt * 16 * DKd + 32, lq, DKd, half);

  v8f acc0 = {}, acc1 = {}, acc2 = {}, acc3 = {};
  float rm[8], rl[8];
#pragma unroll
  for (int r = 0; r < 8; ++r) { rm[r] = -3.0e38f; rl[r] = 0.0f; }

  int nkb_own = qt / 2 + 1;              // this wave's causal key-block count
  int nkb_max = (qbase + 7) / 2 + 1;     // block-uniform trip count

  // prologue: stage key-block 0 (each thread: 1x16B of K, 1x16B of V)
  cp16_g2l(&ksh[0][tid >> 3][(tid & 7) * 8],
           Kb + (size_t)(tid >> 3) * DKd + (tid & 7) * 8);
  cp16_g2l(&vsh[0][tid >> 2][(tid & 3) * 8],
           Vb + (size_t)(tid >> 2) * Sd + (tid & 3) * 8);

  for (int kb = 0; kb < nkb_max; ++kb) {
    int cur = kb & 1;
    wait_async_copies();
    __syncthreads();
    if (kb + 1 < nkb_max) {
      cp16_g2l(&ksh[cur ^ 1][tid >> 3][(tid & 7) * 8],
               Kb + (size_t)((kb + 1) * 32 + (tid >> 3)) * DKd + (tid & 7) * 8);
      cp16_g2l(&vsh[cur ^ 1][tid >> 2][(tid & 3) * 8],
               Vb + (size_t)(tid >> 2) * Sd + (kb + 1) * 32 + (tid & 3) * 8);
    }
    if (kb < nkb_own) {
      // ---- S = Q K^T from LDS K tile ----
      v8f s0 = {}, s1 = {};
      v16bf b00 = load_frag16(&ksh[cur][0][0],   lq, 64, half);
      v16bf b01 = load_frag16(&ksh[cur][0][32],  lq, 64, half);
      v16bf b10 = load_frag16(&ksh[cur][16][0],  lq, 64, half);
      v16bf b11 = load_frag16(&ksh[cur][16][32], lq, 64, half);
      s0 = __builtin_amdgcn_wmma_f32_16x16x32_bf16(false, qa0, false, b00, (short)0, s0, false, false);
      s0 = __builtin_amdgcn_wmma_f32_16x16x32_bf16(false, qa1, false, b01, (short)0, s0, false, false);
      s1 = __builtin_amdgcn_wmma_f32_16x16x32_bf16(false, qa0, false, b10, (short)0, s1, false, false);
      s1 = __builtin_amdgcn_wmma_f32_16x16x32_bf16(false, qa1, false, b11, (short)0, s1, false, false);
      // ---- scale, causal mask, online softmax ----
      float p0[8], p1[8];
#pragma unroll
      for (int r = 0; r < 8; ++r) {
        int m  = qt * 16 + r + half * 8;
        int n0 = kb * 32 + lq;
        int n1 = n0 + 16;
        float v0 = (n0 <= m) ? s0[r] * 0.125f : -1.0e30f;
        float v1 = (n1 <= m) ? s1[r] * 0.125f : -1.0e30f;
        float t = fmaxf(v0, v1);
        t = fmaxf(t, __shfl_xor(t, 1));
        t = fmaxf(t, __shfl_xor(t, 2));
        t = fmaxf(t, __shfl_xor(t, 4));
        t = fmaxf(t, __shfl_xor(t, 8));
        float nm   = fmaxf(rm[r], t);
        float corr = __expf(rm[r] - nm);
        rm[r] = nm;
        p0[r] = __expf(v0 - nm);
        p1[r] = __expf(v1 - nm);
        float ps = p0[r] + p1[r];
        ps += __shfl_xor(ps, 1);
        ps += __shfl_xor(ps, 2);
        ps += __shfl_xor(ps, 4);
        ps += __shfl_xor(ps, 8);
        rl[r] = rl[r] * corr + ps;
        acc0[r] *= corr; acc1[r] *= corr; acc2[r] *= corr; acc3[r] *= corr;
      }
      // ---- P (C-layout) -> LDS -> A-layout fragment ----
#pragma unroll
      for (int r = 0; r < 8; ++r) {
        pbuf[wid][r + half * 8][lq]      = f2bf(p0[r]);
        pbuf[wid][r + half * 8][lq + 16] = f2bf(p1[r]);
      }
      __builtin_amdgcn_wave_barrier();  // keep DS store->load ordering in wave
      v16bf pa = load_frag16(&pbuf[wid][0][0], lq, 32, half);
      // ---- O += P @ V from LDS V^T tile ----
      v16bf vb0 = load_frag16(&vsh[cur][0][0],  lq, 32, half);
      v16bf vb1 = load_frag16(&vsh[cur][16][0], lq, 32, half);
      v16bf vb2 = load_frag16(&vsh[cur][32][0], lq, 32, half);
      v16bf vb3 = load_frag16(&vsh[cur][48][0], lq, 32, half);
      acc0 = __builtin_amdgcn_wmma_f32_16x16x32_bf16(false, pa, false, vb0, (short)0, acc0, false, false);
      acc1 = __builtin_amdgcn_wmma_f32_16x16x32_bf16(false, pa, false, vb1, (short)0, acc1, false, false);
      acc2 = __builtin_amdgcn_wmma_f32_16x16x32_bf16(false, pa, false, vb2, (short)0, acc2, false, false);
      acc3 = __builtin_amdgcn_wmma_f32_16x16x32_bf16(false, pa, false, vb3, (short)0, acc3, false, false);
    }
    __syncthreads();
  }

  // ---- epilogue: normalize by row sums, store [B,S,D] bf16 ----
  int bb = bh / Hd, h = bh % Hd;
#pragma unroll
  for (int r = 0; r < 8; ++r) {
    float inv = 1.0f / rl[r];
    int m = qt * 16 + r + half * 8;
    size_t row = (size_t)(bb * Sd + m) * Dd + h * DKd;
    O[row + 0 * 16 + lq] = f2bf(acc0[r] * inv);
    O[row + 1 * 16 + lq] = f2bf(acc1[r] * inv);
    O[row + 2 * 16 + lq] = f2bf(acc2[r] * inv);
    O[row + 3 * 16 + lq] = f2bf(acc3[r] * inv);
  }
}

// ---------------- Output projection: out = A @ Wo^T + b ----------------
__global__ __launch_bounds__(256) void out_proj_kernel(
    const __bf16* __restrict__ A,    // [Md, Dd] bf16
    const __bf16* __restrict__ W,    // [Dd, Dd] bf16
    const float* __restrict__ bias,  // [Dd] f32
    float* __restrict__ out) {       // [Md, Dd] f32
  __shared__ __bf16 bsh[2][16][32];
  int tid  = threadIdx.x;
  int lane = tid & 31;
  int wid  = tid >> 5;
  int half = lane >> 4;
  int lq   = lane & 15;
  int mt = blockIdx.y * 8 + wid;
  int nt = blockIdx.x;

  const __bf16* arow = A + (size_t)(mt * 16) * Dd;
  const __bf16* brow = W + (size_t)(nt * 16) * Dd;

  if (tid < 64)
    cp16_g2l(&bsh[0][tid >> 2][(tid & 3) * 8], brow + (size_t)(tid >> 2) * Dd + (tid & 3) * 8);

  v8f acc = {};
  for (int kk = 0; kk < Dd; kk += 32) {
    int cur = (kk >> 5) & 1;
    wait_async_copies();
    __syncthreads();
    if (kk + 32 < Dd && tid < 64)
      cp16_g2l(&bsh[cur ^ 1][tid >> 2][(tid & 3) * 8],
               brow + kk + 32 + (size_t)(tid >> 2) * Dd + (tid & 3) * 8);
    __builtin_prefetch(arow + kk + 128, 0, 3);
    v16bf a = load_frag16(arow + kk, lq, Dd, half);
    v16bf b = load_frag16(&bsh[cur][0][0], lq, 32, half);
    acc = __builtin_amdgcn_wmma_f32_16x16x32_bf16(false, a, false, b,
                                                  (short)0, acc, false, false);
    __syncthreads();
  }
#pragma unroll
  for (int r = 0; r < 8; ++r) {
    int m = mt * 16 + r + half * 8;
    int n = nt * 16 + lq;
    out[(size_t)m * Dd + n] = acc[r] + bias[n];
  }
}

extern "C" void kernel_launch(void* const* d_in, const int* in_sizes, int n_in,
                              void* d_out, int out_size, void* d_ws, size_t ws_size,
                              hipStream_t stream) {
  const float* x  = (const float*)d_in[0];
  const float* wq = (const float*)d_in[1];
  const float* wk = (const float*)d_in[2];
  const float* wv = (const float*)d_in[3];
  const float* wo = (const float*)d_in[4];
  const float* bo = (const float*)d_in[5];

  char* ws = (char*)d_ws;
  size_t off = 0;
  __bf16* xb   = (__bf16*)(ws + off); off += (size_t)Md * Dd * 2;     // 16 MiB
  __bf16* wqb  = (__bf16*)(ws + off); off += (size_t)Dd * Dd * 2;     //  2 MiB
  __bf16* wkb  = (__bf16*)(ws + off); off += (size_t)Dd * Dd * 2;
  __bf16* wvb  = (__bf16*)(ws + off); off += (size_t)Dd * Dd * 2;
  __bf16* wob  = (__bf16*)(ws + off); off += (size_t)Dd * Dd * 2;
  __bf16* Qh   = (__bf16*)(ws + off); off += (size_t)Md * Dd * 2;     // [B,H,S,DK]
  __bf16* Kh   = (__bf16*)(ws + off); off += (size_t)Md * Dd * 2;     // [B,H,S,DK]
  __bf16* Vt   = (__bf16*)(ws + off); off += (size_t)Md * Dd * 2;     // [B,H,DK,S]
  __bf16* attn = (__bf16*)(ws + off); off += (size_t)Md * Dd * 2;     // [B,S,D]

  // 1) convert inputs to bf16
  cvt_f32_bf16_kernel<<<4096, 256, 0, stream>>>(x,  xb,  Md * Dd);
  cvt_f32_bf16_kernel<<<1024, 256, 0, stream>>>(wq, wqb, Dd * Dd);
  cvt_f32_bf16_kernel<<<1024, 256, 0, stream>>>(wk, wkb, Dd * Dd);
  cvt_f32_bf16_kernel<<<1024, 256, 0, stream>>>(wv, wvb, Dd * Dd);
  cvt_f32_bf16_kernel<<<1024, 256, 0, stream>>>(wo, wob, Dd * Dd);

  // 2) Q/K/V projections (WMMA GEMMs with async-LDS B staging)
  dim3 g(Dd / 16, Md / 128);  // 64 x 64 blocks, 8 waves each
  qkv_gemm_kernel<<<g, 256, 0, stream>>>(xb, wqb, Qh, 0);
  qkv_gemm_kernel<<<g, 256, 0, stream>>>(xb, wkb, Kh, 1);
  qkv_gemm_kernel<<<g, 256, 0, stream>>>(xb, wvb, Vt, 2);

  // 3) causal flash attention (8 q-tiles per block share LDS-staged K/V)
  flash_attn_kernel<<<(Bd * Hd * (Sd / 16)) / 8, 256, 0, stream>>>(Qh, Kh, Vt, attn);

  // 4) output projection + bias -> f32 out
  out_proj_kernel<<<g, 256, 0, stream>>>(attn, wob, bo, (float*)d_out);
}
